// DenoiseLSTM_20031727468927
// MI455X (gfx1250) — matrix operand order repaired and checked by reading, not compiled, not run on hardware
//
#include <hip/hip_runtime.h>

#define B_  64
#define L_  64
#define T_  32
#define V_  32000
#define DE_ 128
#define DH_ 256
#define DD_ 512

typedef __attribute__((ext_vector_type(16))) __bf16        v16bf;
typedef __attribute__((ext_vector_type(8)))  float         v8f;
typedef __attribute__((ext_vector_type(8)))  unsigned short v8us;
typedef __attribute__((ext_vector_type(16))) unsigned short v16us;

__device__ __forceinline__ unsigned short f2bf(float f) {
  unsigned u = __float_as_uint(f);
  u += 0x7fffu + ((u >> 16) & 1u);          // round-to-nearest-even
  return (unsigned short)(u >> 16);
}
__device__ __forceinline__ float sigf(float x) { return 1.f / (1.f + __expf(-x)); }

// A-tile 16x32 bf16 (ISA 7.12.2): lane = m + 16*g ; element e -> k = e + 8*((e>>3)+g)
__device__ __forceinline__ v16bf load_a(const unsigned short* p, int ld, int row, int k0, int g) {
  const unsigned short* q = p + (size_t)row * ld + k0 + 8 * g;
  v8us lo = *(const v8us*)(q);
  v8us hi = *(const v8us*)(q + 16);
  v16us w = __builtin_shufflevector(lo, hi, 0,1,2,3,4,5,6,7,8,9,10,11,12,13,14,15);
  return __builtin_bit_cast(v16bf, w);
}
// B-tile 32x16 bf16: lane n holds column n of B == row n of W[N][K]; k = e + 8*((e>>3)+g) (same as A)
__device__ __forceinline__ v16bf load_b(const unsigned short* p, int ld, int row, int k0, int g) {
  const unsigned short* q = p + (size_t)row * ld + k0 + 8 * g;
  v8us lo = *(const v8us*)(q);
  v8us hi = *(const v8us*)(q + 16);
  v16us w = __builtin_shufflevector(lo, hi, 0,1,2,3,4,5,6,7,8,9,10,11,12,13,14,15);
  return __builtin_bit_cast(v16bf, w);
}

// ---------------- generic bf16 WMMA GEMM, 16x64 per wave (A fragment reused 4x):
//  C[M,N](f32) = A[M,K] @ W[N,K]^T (+bias)(+tanh) ; N % 64 == 0, M % 16 == 0, K % 32 == 0
__global__ void gemm_bf16(const unsigned short* __restrict__ A, int lda,
                          const unsigned short* __restrict__ Bw, int ldb,
                          float* __restrict__ C, long long ldc,
                          const float* __restrict__ bias,
                          int M, int N, int K, int act)
{
  int lane = threadIdx.x & 31;
  int m = lane & 15, g = lane >> 4;
  int wid = (int)((blockIdx.x * blockDim.x + threadIdx.x) >> 5);
  int nw  = (int)((gridDim.x * blockDim.x) >> 5);
  int tM = M >> 4, tN4 = N >> 6;
  for (int t = wid; t < tM * tN4; t += nw) {
    int ti = t / tN4, tj = t - ti * tN4;
    int col0 = tj * 64 + m;
    v8f acc[4];
#pragma unroll
    for (int jj = 0; jj < 4; ++jj) {
      float bv = bias ? bias[col0 + jj * 16] : 0.f;
#pragma unroll
      for (int e = 0; e < 8; ++e) acc[jj][e] = bv;
    }
    for (int k0 = 0; k0 < K; k0 += 32) {
      v16bf a = load_a(A, lda, ti * 16 + m, k0, g);
#pragma unroll
      for (int jj = 0; jj < 4; ++jj) {
        v16bf b = load_b(Bw, ldb, col0 + jj * 16, k0, g);
        acc[jj] = __builtin_amdgcn_wmma_f32_16x16x32_bf16(false, a, false, b, (short)0, acc[jj], false, false);
      }
    }
    if (act == 1) {
#pragma unroll
      for (int jj = 0; jj < 4; ++jj)
#pragma unroll
        for (int e = 0; e < 8; ++e) acc[jj][e] = tanhf(acc[jj][e]);
    }
#pragma unroll
    for (int jj = 0; jj < 4; ++jj)
#pragma unroll
      for (int e = 0; e < 8; ++e) {
        int r = ti * 16 + e + 8 * g;
        C[(long long)r * ldc + col0 + jj * 16] = acc[jj][e];
      }
  }
}

// ---------------- weight prep kernels
__global__ void cvt_bf16(unsigned short* dst, const float* src, long long n) {
  long long i = (long long)blockIdx.x * blockDim.x + threadIdx.x;
  if (i < n) dst[i] = f2bf(src[i]);
}
// gate-permuted conversion: new row rn = unit*4 + gate; source row = gate*units + unit
__global__ void cvt_perm_gate(unsigned short* dst, const float* src, float* bdst,
                              const float* bsrc, int units, int K) {
  long long n = (long long)units * 4 * K;
  long long i = (long long)blockIdx.x * blockDim.x + threadIdx.x;
  if (i >= n) return;
  int rn = (int)(i / K), c = (int)(i - (long long)rn * K);
  int u = rn >> 2, gi = rn & 3;
  dst[i] = f2bf(src[((size_t)gi * units + u) * K + c]);
  if (c == 0 && bdst) bdst[rn] = bsrc[gi * units + u];
}
// decoder combined [Wih|Whh] -> [2048][640], gate-permuted rows + permuted bias
__global__ void build_decW(unsigned short* dst, float* bdst, const float* Wih,
                           const float* Whh, const float* bsrc) {
  long long n = (long long)2048 * 640;
  long long i = (long long)blockIdx.x * blockDim.x + threadIdx.x;
  if (i >= n) return;
  int rn = (int)(i / 640), c = (int)(i - (long long)rn * 640);
  int u = rn >> 2, gi = rn & 3;
  int srow = gi * 512 + u;
  float v = (c < 128) ? Wih[(size_t)srow * 128 + c] : Whh[(size_t)srow * 512 + (c - 128)];
  dst[i] = f2bf(v);
  if (c == 0) bdst[rn] = bsrc[srow];
}
__global__ void build_xs(unsigned short* xs, const unsigned short* tok, const int* nx) {
  int lb = blockIdx.x;            // l*64 + b
  int l = lb >> 6, b = lb & 63;
  int tk = nx[b * 64 + l];
  xs[(size_t)lb * 128 + threadIdx.x] = tok[(size_t)tk * 128 + threadIdx.x];
}
__global__ void zero_init(unsigned short* a, unsigned short* b, float* c, int n) {
  int i = blockIdx.x * blockDim.x + threadIdx.x;
  if (i < n) { a[i] = 0; b[i] = 0; c[i] = 0.f; }
}

// ---------------- fused encoder recurrent step (gate-permuted cols; barrier-free pointwise)
// wave computes a 16x64 strip (4 units x 4 gates x 4 WMMA tiles) then applies the LSTM cell
__global__ void enc_step(const unsigned short* __restrict__ WhhAll, // [2][1024p][256]
                         const float* __restrict__ Xpre,            // [2][4096][1024p]
                         const unsigned short* __restrict__ hR,     // [2][64][256]
                         unsigned short* __restrict__ hW,
                         float* __restrict__ cbuf,                  // [2][64][256]
                         unsigned short* __restrict__ membf,        // [64][64][512]
                         unsigned short* __restrict__ hfb,          // [64][512]
                         int s)
{
  __shared__ float lsl[8][16][64];
  int dir = blockIdx.x >> 2;
  int rowBase = (blockIdx.x & 3) << 4;
  int posIn = dir ? (63 - s) : s;
  const unsigned short* Whh = WhhAll + (size_t)dir * 1024 * 256;
  const unsigned short* hb  = hR + (size_t)dir * 64 * 256;
  const float* xp = Xpre + ((size_t)dir * 4096 + (size_t)posIn * 64) * 1024;
  int lane = threadIdx.x & 31, m = lane & 15, g = lane >> 4;
  int wave = threadIdx.x >> 5;
  float (*ls)[64] = lsl[wave];
  for (int st = wave * 2; st < wave * 2 + 2; ++st) {      // supertiles 0..15, 16 units each
    v8f acc[4];
#pragma unroll
    for (int jj = 0; jj < 4; ++jj)
#pragma unroll
      for (int e = 0; e < 8; ++e) acc[jj][e] = 0.f;
#pragma unroll
    for (int k0 = 0; k0 < 256; k0 += 32) {
      v16bf a = load_a(hb, 256, rowBase + m, k0, g);
#pragma unroll
      for (int jj = 0; jj < 4; ++jj) {
        v16bf b = load_b(Whh, 256, st * 64 + jj * 16 + m, k0, g);
        acc[jj] = __builtin_amdgcn_wmma_f32_16x16x32_bf16(false, a, false, b, (short)0, acc[jj], false, false);
      }
    }
#pragma unroll
    for (int jj = 0; jj < 4; ++jj)
#pragma unroll
      for (int e = 0; e < 8; ++e) {
        int r = e + 8 * g;
        ls[r][jj * 16 + m] = acc[jj][e] + xp[(size_t)(rowBase + r) * 1024 + st * 64 + jj * 16 + m];
      }
    // per-wave pointwise: 16 rows x 16 units (local col = uu*4 + gate)
#pragma unroll
    for (int it = 0; it < 8; ++it) {
      int item = lane * 8 + it;
      int r = item >> 4, uu = item & 15;
      int u = st * 16 + uu;
      int b = rowBase + r;
      float ig = ls[r][uu * 4 + 0], fg = ls[r][uu * 4 + 1];
      float gg = ls[r][uu * 4 + 2], og = ls[r][uu * 4 + 3];
      size_t ci = ((size_t)dir * 64 + b) * 256 + u;
      float c = sigf(fg) * cbuf[ci] + sigf(ig) * tanhf(gg);
      cbuf[ci] = c;
      float h = sigf(og) * tanhf(c);
      unsigned short hv = f2bf(h);
      hW[ci] = hv;
      membf[((size_t)posIn * 64 + b) * 512 + dir * 256 + u] = hv;
      if (s == 63) hfb[(size_t)b * 512 + dir * 256 + u] = hv;
    }
  }
}

// ---------------- fused decoder LSTM cell (gates + pointwise), 16x64 strips per wave
__global__ void dec_cell(const unsigned short* __restrict__ decW, // [2048p][640]
                         const float* __restrict__ decbp,         // [2048p]
                         const unsigned short* __restrict__ cellA,// [64][640] = [xe|h]
                         float* __restrict__ cdec, float* __restrict__ hcell,
                         unsigned short* __restrict__ hud)
{
  __shared__ float lsl[8][16][64];
  int rowBase = blockIdx.x << 4;
  int lane = threadIdx.x & 31, m = lane & 15, g = lane >> 4;
  int wave = threadIdx.x >> 5;
  float (*ls)[64] = lsl[wave];
  for (int st = wave * 4; st < wave * 4 + 4; ++st) {      // supertiles 0..31, 16 units each
    v8f acc[4];
#pragma unroll
    for (int jj = 0; jj < 4; ++jj) {
      float bv = decbp[st * 64 + jj * 16 + m];
#pragma unroll
      for (int e = 0; e < 8; ++e) acc[jj][e] = bv;
    }
    for (int k0 = 0; k0 < 640; k0 += 32) {
      v16bf a = load_a(cellA, 640, rowBase + m, k0, g);
#pragma unroll
      for (int jj = 0; jj < 4; ++jj) {
        v16bf b = load_b(decW, 640, st * 64 + jj * 16 + m, k0, g);
        acc[jj] = __builtin_amdgcn_wmma_f32_16x16x32_bf16(false, a, false, b, (short)0, acc[jj], false, false);
      }
    }
#pragma unroll
    for (int jj = 0; jj < 4; ++jj)
#pragma unroll
      for (int e = 0; e < 8; ++e) ls[e + 8 * g][jj * 16 + m] = acc[jj][e];
#pragma unroll
    for (int it = 0; it < 8; ++it) {
      int item = lane * 8 + it;
      int r = item >> 4, uu = item & 15;
      int u = st * 16 + uu;
      int b = rowBase + r;
      float ig = ls[r][uu * 4 + 0], fg = ls[r][uu * 4 + 1];
      float gg = ls[r][uu * 4 + 2], og = ls[r][uu * 4 + 3];
      size_t ci = (size_t)b * 512 + u;
      float c = sigf(fg) * cdec[ci] + sigf(ig) * tanhf(gg);
      cdec[ci] = c;
      float h = sigf(og) * tanhf(c);
      hcell[ci] = h;
      hud[ci] = f2bf(h);
    }
  }
}

// ---------------- single-query multi-head attention, one block per (b,h)
__global__ void attn_kernel(const float* __restrict__ q, const float* __restrict__ Km,
                            const float* __restrict__ Vm, unsigned short* __restrict__ ctx)
{
  int b = blockIdx.x, h = blockIdx.y, l = threadIdx.x;   // 64 threads
  __shared__ float sc[64], red[64];
  const float* qp = q + (size_t)b * 512 + h * 64;
  const float* kp = Km + ((size_t)l * 64 + b) * 512 + h * 64;
  float s = 0.f;
#pragma unroll 8
  for (int d = 0; d < 64; ++d) s += qp[d] * kp[d];
  s *= 0.125f;                                           // 1/sqrt(64)
  sc[l] = s; red[l] = s; __syncthreads();
  for (int st = 32; st > 0; st >>= 1) { if (l < st) red[l] = fmaxf(red[l], red[l + st]); __syncthreads(); }
  float mx = red[0]; __syncthreads();
  float e = __expf(s - mx);
  sc[l] = e; red[l] = e; __syncthreads();
  for (int st = 32; st > 0; st >>= 1) { if (l < st) red[l] += red[l + st]; __syncthreads(); }
  float inv = 1.f / red[0];
  __syncthreads();
  float acc = 0.f;
  for (int j = 0; j < 64; ++j) acc += sc[j] * Vm[((size_t)j * 64 + b) * 512 + h * 64 + l];
  ctx[(size_t)b * 512 + h * 64 + l] = f2bf(acc * inv);
}

// ---------------- residual + LayerNorm -> writes post-LN h (bf16) into cellA[:,128:640]
__global__ void ln_res(const float* __restrict__ hcell, const float* __restrict__ aout,
                       const float* __restrict__ gam, const float* __restrict__ bet,
                       unsigned short* __restrict__ cellA)
{
  int b = blockIdx.x, tid = threadIdx.x;                 // 256 threads
  __shared__ float red[256];
  float y0 = hcell[(size_t)b * 512 + tid]       + aout[(size_t)b * 512 + tid];
  float y1 = hcell[(size_t)b * 512 + 256 + tid] + aout[(size_t)b * 512 + 256 + tid];
  red[tid] = y0 + y1; __syncthreads();
  for (int s = 128; s > 0; s >>= 1) { if (tid < s) red[tid] += red[tid + s]; __syncthreads(); }
  float mu = red[0] * (1.f / 512.f); __syncthreads();
  float d0 = y0 - mu, d1 = y1 - mu;
  red[tid] = d0 * d0 + d1 * d1; __syncthreads();
  for (int s = 128; s > 0; s >>= 1) { if (tid < s) red[tid] += red[tid + s]; __syncthreads(); }
  float rstd = rsqrtf(red[0] * (1.f / 512.f) + 1e-5f);
  cellA[(size_t)b * 640 + 128 + tid]       = f2bf(d0 * rstd * gam[tid] + bet[tid]);
  cellA[(size_t)b * 640 + 128 + 256 + tid] = f2bf(d1 * rstd * gam[tid + 256] + bet[tid + 256]);
}

// ---------------- greedy argmax over V + embed gather of next token into cellA[:,0:128]
__global__ void argmax_gather(const float* __restrict__ logits, long long rowStride,
                              const unsigned short* __restrict__ tok,
                              unsigned short* __restrict__ cellA)
{
  int b = blockIdx.x, tid = threadIdx.x;                 // 256 threads
  __shared__ float rv[256]; __shared__ int ri[256];
  const float* row = logits + (long long)b * rowStride;
  float bv = -3.4e38f; int bi = 0;
  for (int v = tid; v < V_; v += 256) { float x = row[v]; if (x > bv) { bv = x; bi = v; } }
  rv[tid] = bv; ri[tid] = bi; __syncthreads();
  for (int s = 128; s > 0; s >>= 1) {
    if (tid < s) {
      if (rv[tid + s] > rv[tid] || (rv[tid + s] == rv[tid] && ri[tid + s] < ri[tid])) {
        rv[tid] = rv[tid + s]; ri[tid] = ri[tid + s];
      }
    }
    __syncthreads();
  }
  int best = ri[0];
  if (tid < 128) cellA[(size_t)b * 640 + tid] = tok[(size_t)best * 128 + tid];
}

__global__ void dec_init(float* __restrict__ cdec, unsigned short* __restrict__ cellA,
                         const float* __restrict__ style, const int* __restrict__ label,
                         const float* __restrict__ start_emb, const float* __restrict__ h0f)
{
  int b = blockIdx.x, u = threadIdx.x;                   // 512 threads
  cdec[(size_t)b * 512 + u] = style[(size_t)label[b] * 512 + u];
  cellA[(size_t)b * 640 + 128 + u] = f2bf(h0f[(size_t)b * 512 + u]);
  if (u < 128) cellA[(size_t)b * 640 + u] = f2bf(start_emb[u]);
}

extern "C" void kernel_launch(void* const* d_in, const int* in_sizes, int n_in,
                              void* d_out, int out_size, void* d_ws, size_t ws_size,
                              hipStream_t stream) {
  (void)in_sizes; (void)n_in; (void)out_size; (void)ws_size;
  const int*   nx        = (const int*)d_in[0];
  const int*   label     = (const int*)d_in[1];
  const float* start_emb = (const float*)d_in[2];
  const float* tok_emb   = (const float*)d_in[3];
  const float* style_emb = (const float*)d_in[4];
  const float* enc_Wih_f = (const float*)d_in[5];
  const float* enc_Whh_f = (const float*)d_in[6];
  const float* enc_b_f   = (const float*)d_in[7];
  const float* enc_Wih_b = (const float*)d_in[8];
  const float* enc_Whh_b = (const float*)d_in[9];
  const float* enc_b_b   = (const float*)d_in[10];
  const float* transfer_W= (const float*)d_in[11];
  const float* dec_Wih   = (const float*)d_in[12];
  const float* dec_Whh   = (const float*)d_in[13];
  const float* dec_b     = (const float*)d_in[14];
  const float* attn_in_w = (const float*)d_in[15];
  const float* attn_in_b = (const float*)d_in[16];
  const float* attn_out_w= (const float*)d_in[17];
  const float* attn_out_b= (const float*)d_in[18];
  const float* proj_W    = (const float*)d_in[19];
  const float* proj_b    = (const float*)d_in[20];
  const float* ln_g      = (const float*)d_in[21];
  const float* ln_bb     = (const float*)d_in[22];
  float* out = (float*)d_out;

  char* base = (char*)d_ws; size_t off = 0;
  auto alloc = [&](size_t bytes) -> void* {
    off = (off + 255) & ~(size_t)255;
    void* p = base + off; off += bytes; return p;
  };
  unsigned short* tokbf   = (unsigned short*)alloc((size_t)V_ * DE_ * 2);
  unsigned short* wihp    = (unsigned short*)alloc((size_t)2 * 1024 * DE_ * 2);
  unsigned short* whhp    = (unsigned short*)alloc((size_t)2 * 1024 * DH_ * 2);
  float*          encbp   = (float*)alloc((size_t)2 * 1024 * 4);
  unsigned short* transbf = (unsigned short*)alloc((size_t)DD_ * DD_ * 2);
  unsigned short* decWbf  = (unsigned short*)alloc((size_t)2048 * 640 * 2);
  float*          decbp   = (float*)alloc((size_t)2048 * 4);
  unsigned short* attnWbf = (unsigned short*)alloc((size_t)1536 * DD_ * 2);
  unsigned short* attnOWbf= (unsigned short*)alloc((size_t)DD_ * DD_ * 2);
  unsigned short* projWbf = (unsigned short*)alloc((size_t)V_ * DD_ * 2);
  unsigned short* xsbf    = (unsigned short*)alloc((size_t)L_ * B_ * DE_ * 2);
  unsigned short* membf   = (unsigned short*)alloc((size_t)L_ * B_ * DD_ * 2);
  unsigned short* hbfA    = (unsigned short*)alloc((size_t)2 * B_ * DH_ * 2);
  unsigned short* hbfB    = (unsigned short*)alloc((size_t)2 * B_ * DH_ * 2);
  unsigned short* hfbbf   = (unsigned short*)alloc((size_t)B_ * DD_ * 2);
  unsigned short* cellA   = (unsigned short*)alloc((size_t)B_ * 640 * 2);
  unsigned short* hudbf   = (unsigned short*)alloc((size_t)B_ * DD_ * 2);
  unsigned short* ctxbf   = (unsigned short*)alloc((size_t)B_ * DD_ * 2);
  float* Xpre  = (float*)alloc((size_t)2 * L_ * B_ * 1024 * 4);
  float* cenc  = (float*)alloc((size_t)2 * B_ * DH_ * 4);
  float* Kmat  = (float*)alloc((size_t)L_ * B_ * DD_ * 4);
  float* Vmat  = (float*)alloc((size_t)L_ * B_ * DD_ * 4);
  float* h0f   = (float*)alloc((size_t)B_ * DD_ * 4);
  float* cdec  = (float*)alloc((size_t)B_ * DD_ * 4);
  float* hcell = (float*)alloc((size_t)B_ * DD_ * 4);
  float* qbuf  = (float*)alloc((size_t)B_ * DD_ * 4);
  float* aoutb = (float*)alloc((size_t)B_ * DD_ * 4);

  auto nb = [](long long n) { return (unsigned)((n + 255) / 256); };

  // ---- weight prep (bf16 + gate permutation)
  cvt_bf16<<<nb((long long)V_ * DE_), 256, 0, stream>>>(tokbf, tok_emb, (long long)V_ * DE_);
  cvt_perm_gate<<<nb(1024LL * DE_), 256, 0, stream>>>(wihp, enc_Wih_f, encbp, enc_b_f, 256, 128);
  cvt_perm_gate<<<nb(1024LL * DE_), 256, 0, stream>>>(wihp + 1024 * DE_, enc_Wih_b, encbp + 1024, enc_b_b, 256, 128);
  cvt_perm_gate<<<nb(1024LL * DH_), 256, 0, stream>>>(whhp, enc_Whh_f, nullptr, nullptr, 256, 256);
  cvt_perm_gate<<<nb(1024LL * DH_), 256, 0, stream>>>(whhp + 1024 * DH_, enc_Whh_b, nullptr, nullptr, 256, 256);
  cvt_bf16<<<nb((long long)DD_ * DD_), 256, 0, stream>>>(transbf, transfer_W, (long long)DD_ * DD_);
  build_decW<<<nb(2048LL * 640), 256, 0, stream>>>(decWbf, decbp, dec_Wih, dec_Whh, dec_b);
  cvt_bf16<<<nb(1536LL * DD_), 256, 0, stream>>>(attnWbf, attn_in_w, 1536LL * DD_);
  cvt_bf16<<<nb((long long)DD_ * DD_), 256, 0, stream>>>(attnOWbf, attn_out_w, (long long)DD_ * DD_);
  cvt_bf16<<<nb((long long)V_ * DD_), 256, 0, stream>>>(projWbf, proj_W, (long long)V_ * DD_);

  // ---- embeddings + encoder non-recurrent GEMMs (all timesteps at once)
  build_xs<<<L_ * B_, DE_, 0, stream>>>(xsbf, tokbf, nx);
  gemm_bf16<<<512, 256, 0, stream>>>(xsbf, 128, wihp, 128, Xpre, 1024, encbp, 4096, 1024, 128, 0);
  gemm_bf16<<<512, 256, 0, stream>>>(xsbf, 128, wihp + 1024 * DE_, 128,
                                     Xpre + (size_t)4096 * 1024, 1024, encbp + 1024, 4096, 1024, 128, 0);
  zero_init<<<nb(2LL * B_ * DH_), 256, 0, stream>>>(hbfA, hbfB, cenc, 2 * B_ * DH_);

  // ---- bidirectional LSTM recurrence (both directions per launch; double-buffered h)
  for (int s = 0; s < L_; ++s) {
    const unsigned short* hR = (s & 1) ? hbfB : hbfA;
    unsigned short*       hW = (s & 1) ? hbfA : hbfB;
    enc_step<<<8, 256, 0, stream>>>(whhp, Xpre, hR, hW, cenc, membf, hfbbf, s);
  }

  // ---- h0 = tanh([hf|hb] @ transfer_W^T), K/V precompute
  gemm_bf16<<<4, 256, 0, stream>>>(hfbbf, 512, transbf, 512, h0f, 512, nullptr, 64, 512, 512, 1);
  gemm_bf16<<<256, 256, 0, stream>>>(membf, 512, attnWbf + (size_t)512 * 512, 512,
                                     Kmat, 512, attn_in_b + 512, 4096, 512, 512, 0);
  gemm_bf16<<<256, 256, 0, stream>>>(membf, 512, attnWbf + (size_t)1024 * 512, 512,
                                     Vmat, 512, attn_in_b + 1024, 4096, 512, 512, 0);
  dec_init<<<64, 512, 0, stream>>>(cdec, cellA, style_emb, label, start_emb, h0f);

  // ---- greedy decode loop
  const long long rowStride = (long long)T_ * V_;
  for (int t = 0; t < T_; ++t) {
    dec_cell<<<4, 256, 0, stream>>>(decWbf, decbp, cellA, cdec, hcell, hudbf);
    gemm_bf16<<<4, 256, 0, stream>>>(hudbf, 512, attnWbf, 512, qbuf, 512, attn_in_b, 64, 512, 512, 0);
    attn_kernel<<<dim3(64, 8), 64, 0, stream>>>(qbuf, Kmat, Vmat, ctxbf);
    gemm_bf16<<<4, 256, 0, stream>>>(ctxbf, 512, attnOWbf, 512, aoutb, 512, attn_out_b, 64, 512, 512, 0);
    ln_res<<<64, 256, 0, stream>>>(hcell, aoutb, ln_g, ln_bb, cellA);
    gemm_bf16<<<256, 256, 0, stream>>>(cellA + 128, 640, projWbf, 512,
                                       out + (long long)t * V_, rowStride, proj_b, 64, V_, 512, 0);
    argmax_gather<<<64, 256, 0, stream>>>(out + (long long)t * V_, rowStride, tokbf, cellA);
  }
}
